// Decoder_40913858461993
// MI455X (gfx1250) — matrix operand order repaired and checked by reading, not compiled, not run on hardware
//
#include <hip/hip_runtime.h>
#include <hip/hip_bf16.h>

// Problem constants (from reference)
#define BB 32
#define LL 80
#define TT 40
#define HH 512
#define VV 30000
#define NSTEP (TT - 1)          // step T-1 is dead in the reference
#define MROWS (NSTEP * BB)      // 1248 rows in the batched logits GEMM
#define MTILES (MROWS / 16)     // 78
#define NROWS_LDS 128           // Wout rows staged in LDS per workgroup

typedef __attribute__((ext_vector_type(16))) __bf16 bf16x16;
typedef __attribute__((ext_vector_type(8)))  __bf16 bf16x8;
typedef __attribute__((ext_vector_type(8)))  float  f32x8;

// ---------------------------------------------------------------------------
// WMMA helpers (CDNA5 gfx1250, wave32)
// ---------------------------------------------------------------------------
static __device__ __forceinline__ f32x8 wmma_bf16(bf16x16 a, bf16x16 b, f32x8 c) {
  return __builtin_amdgcn_wmma_f32_16x16x32_bf16(false, a, false, b, (short)0, c,
                                                 false, false);
}

// A-matrix 16x32 bf16 loader, row-major [M][K], leading dim ld.
// Lane l<16: M=m0+l, K = k0+0..15 contiguous ; lanes 16-31: K=k0+16..31.
static __device__ __forceinline__ bf16x16 load_A(const __bf16* __restrict__ A,
                                                 int ld, int m0, int k0) {
  const int lane = threadIdx.x & 31;
  const int m  = m0 + (lane & 15);
  const int kb = k0 + ((lane >> 4) << 4);
  const __bf16* p = A + (size_t)m * ld + kb;
  bf16x8 lo = *(const bf16x8*)(p);
  bf16x8 hi = *(const bf16x8*)(p + 8);
  bf16x16 v;
#pragma unroll
  for (int i = 0; i < 8; ++i) { v[i] = lo[i]; v[i + 8] = hi[i]; }
  return v;
}

// A tile from a per-lane f32 row pointer (embedding gather), cvt to bf16.
// p already includes the lane's K-phase offset; k0 is the K-tile base.
static __device__ __forceinline__ bf16x16 load_A_f32ptr(const float* __restrict__ p,
                                                        int k0) {
  f32x8 lo = *(const f32x8*)(p + k0);
  f32x8 hi = *(const f32x8*)(p + k0 + 8);
  bf16x16 v;
#pragma unroll
  for (int i = 0; i < 8; ++i) { v[i] = (__bf16)lo[i]; v[i + 8] = (__bf16)hi[i]; }
  return v;
}

// B-matrix 32x16 loader with B[k][n] = W[n][k] (X @ W^T), W row-major [N][K].
// Lane l<16: N=n0+l, K=k0+0..15 contiguous ; lanes 16-31: K=k0+16..31.
static __device__ __forceinline__ bf16x16 load_Bt(const __bf16* __restrict__ W,
                                                  int ld, int n0, int k0) {
  const int lane = threadIdx.x & 31;
  const int n  = n0 + (lane & 15);
  const int kb = k0 + ((lane >> 4) << 4);
  return *(const bf16x16*)(W + (size_t)n * ld + kb);
}

static __device__ __forceinline__ float sigm(float x) {
  return 1.0f / (1.0f + expf(-x));
}

// ---------------------------------------------------------------------------
// Prep: f32 -> bf16 weight conversion
// ---------------------------------------------------------------------------
__global__ void k_cvt(const float* __restrict__ src, __bf16* __restrict__ dst, int n) {
  for (int i = blockIdx.x * blockDim.x + threadIdx.x; i < n; i += gridDim.x * blockDim.x)
    dst[i] = (__bf16)src[i];
}

// Wcat[n][k] = k<H ? W_ih[n][k] : W_hh[n][k-H]   (n in [0,4H), k in [0,2H))
__global__ void k_build_wcat(const float* __restrict__ Wih, const float* __restrict__ Whh,
                             __bf16* __restrict__ Wcat) {
  int idx = blockIdx.x * blockDim.x + threadIdx.x;
  if (idx >= 4 * HH * 2 * HH) return;
  int n = idx >> 10, k = idx & (2 * HH - 1);
  float v = (k < HH) ? Wih[n * HH + k] : Whh[n * HH + (k - HH)];
  Wcat[idx] = (__bf16)v;
}

// h_ping = bf16(h0) ; c_st = c0
__global__ void k_init(const float* __restrict__ h0, const float* __restrict__ c0,
                       __bf16* __restrict__ h_bf, float* __restrict__ c_st) {
  int idx = blockIdx.x * blockDim.x + threadIdx.x;
  if (idx >= BB * HH) return;
  h_bf[idx] = (__bf16)h0[idx];
  c_st[idx] = c0[idx];
}

// ---------------------------------------------------------------------------
// slot 0: sos_logits = embed_w[SOS] @ Wout^T + bout broadcast; symbols[:,0]=SOS
// ---------------------------------------------------------------------------
__global__ __launch_bounds__(256)
void k_sos(const float* __restrict__ embed_w, const float* __restrict__ Wout,
           const float* __restrict__ bout, float* __restrict__ out) {
  const int w = threadIdx.x >> 5, lane = threadIdx.x & 31;
  const int v = blockIdx.x * 8 + w;
  if (v < VV) {
    const float* er = embed_w + (size_t)1 * HH;  // SOS token row
    const float* wr = Wout + (size_t)v * HH;
    float s = 0.f;
    for (int k = lane; k < HH; k += 32) s += er[k] * wr[k];
#pragma unroll
    for (int m = 16; m > 0; m >>= 1) s += __shfl_xor(s, m, 32);
    if (lane == 0) {
      float val = s + bout[v];
      for (int b = 0; b < BB; ++b)
        out[((size_t)b * TT + 0) * VV + v] = val;
    }
  }
  if (blockIdx.x == 0 && threadIdx.x < BB)
    out[(size_t)BB * TT * VV + (size_t)threadIdx.x * TT + 0] = 1.0f;
}

// ---------------------------------------------------------------------------
// gates = [embed(tok_t) | h_prev] @ Wcat^T + b_ih + b_hh ; LSTM cell update.
// One wave per 16-column tile of H; writes c (f32) and new h (bf16, ping-pong).
// ---------------------------------------------------------------------------
__global__ __launch_bounds__(32)
void k_gates_lstm(const float* __restrict__ embed_w, const int* __restrict__ tok, int t,
                  const __bf16* __restrict__ h_in, const __bf16* __restrict__ Wcat,
                  const float* __restrict__ b_ih, const float* __restrict__ b_hh,
                  float* __restrict__ c_st, __bf16* __restrict__ h_out) {
  const int j = blockIdx.x;  // H columns j*16..j*16+15
  const int lane = threadIdx.x;
  f32x8 acc[4][2];
#pragma unroll
  for (int g = 0; g < 4; ++g) { acc[g][0] = (f32x8){}; acc[g][1] = (f32x8){}; }

  // Per-lane embedding row pointers (hoisted token gather; fused k_embed)
  const int mlo = lane & 15;
  const int kb8 = (lane >> 4) << 4;
  const float* ep0 = embed_w + (size_t)tok[mlo * TT + t] * HH + kb8;
  const float* ep1 = embed_w + (size_t)tok[(16 + mlo) * TT + t] * HH + kb8;

  // x half: K = 0..H-1 of Wcat
#pragma unroll 4
  for (int kt = 0; kt < HH; kt += 32) {
    bf16x16 a0 = load_A_f32ptr(ep0, kt);
    bf16x16 a1 = load_A_f32ptr(ep1, kt);
#pragma unroll
    for (int g = 0; g < 4; ++g) {
      bf16x16 b = load_Bt(Wcat, 2 * HH, g * HH + j * 16, kt);
      acc[g][0] = wmma_bf16(a0, b, acc[g][0]);
      acc[g][1] = wmma_bf16(a1, b, acc[g][1]);
    }
  }
  // h half: K = H..2H-1 of Wcat
#pragma unroll 4
  for (int kt = 0; kt < HH; kt += 32) {
    bf16x16 a0 = load_A(h_in, HH, 0, kt);
    bf16x16 a1 = load_A(h_in, HH, 16, kt);
#pragma unroll
    for (int g = 0; g < 4; ++g) {
      bf16x16 b = load_Bt(Wcat, 2 * HH, g * HH + j * 16, HH + kt);
      acc[g][0] = wmma_bf16(a0, b, acc[g][0]);
      acc[g][1] = wmma_bf16(a1, b, acc[g][1]);
    }
  }

  const int colH = j * 16 + (lane & 15);
  const float bi = b_ih[0 * HH + colH] + b_hh[0 * HH + colH];
  const float bf = b_ih[1 * HH + colH] + b_hh[1 * HH + colH];
  const float bg = b_ih[2 * HH + colH] + b_hh[2 * HH + colH];
  const float bo = b_ih[3 * HH + colH] + b_hh[3 * HH + colH];
#pragma unroll
  for (int mi = 0; mi < 2; ++mi)
#pragma unroll
    for (int r = 0; r < 8; ++r) {
      int row = mi * 16 + ((lane >> 4) << 3) + r;  // batch index
      float iv = acc[0][mi][r] + bi;
      float fv = acc[1][mi][r] + bf;
      float gv = acc[2][mi][r] + bg;
      float ov = acc[3][mi][r] + bo;
      float co = c_st[row * HH + colH];
      float cn = sigm(fv) * co + sigm(iv) * tanhf(gv);
      float hn = sigm(ov) * tanhf(cn);
      c_st[row * HH + colH] = cn;
      h_out[row * HH + colH] = (__bf16)hn;
    }
}

// e = h @ attn_w^T + attn_b
__global__ __launch_bounds__(32)
void k_attn_e(const __bf16* __restrict__ h_bf, const __bf16* __restrict__ attn_bf,
              const float* __restrict__ attn_b, float* __restrict__ e) {
  const int j = blockIdx.x;
  const int lane = threadIdx.x;
  f32x8 acc0 = (f32x8){}, acc1 = (f32x8){};
#pragma unroll 4
  for (int kt = 0; kt < HH; kt += 32) {
    bf16x16 a0 = load_A(h_bf, HH, 0, kt);
    bf16x16 a1 = load_A(h_bf, HH, 16, kt);
    bf16x16 b  = load_Bt(attn_bf, HH, j * 16, kt);
    acc0 = wmma_bf16(a0, b, acc0);
    acc1 = wmma_bf16(a1, b, acc1);
  }
  const int col = j * 16 + (lane & 15);
  const float bias = attn_b[col];
#pragma unroll
  for (int r = 0; r < 8; ++r) {
    int r0 = ((lane >> 4) << 3) + r;
    e[r0 * HH + col]        = acc0[r] + bias;
    e[(16 + r0) * HH + col] = acc1[r] + bias;
  }
}

// scores = e . enc over L, softmax, ctx = A . enc  -> ctx_bf (bf16)
__global__ __launch_bounds__(256)
void k_attn_ctx(const float* __restrict__ e, const float* __restrict__ enc,
                __bf16* __restrict__ ctx_bf) {
  const int b = blockIdx.x;
  const int w = threadIdx.x >> 5, lane = threadIdx.x & 31;
  __shared__ float sc[LL];

  for (int l = w; l < LL; l += 8) {
    const float* po = enc + ((size_t)b * LL + l) * HH;
    const float* pe = e + (size_t)b * HH;
    float s = 0.f;
    for (int k = lane; k < HH; k += 32) s += pe[k] * po[k];
#pragma unroll
    for (int m = 16; m > 0; m >>= 1) s += __shfl_xor(s, m, 32);
    if (lane == 0) sc[l] = s;
  }
  __syncthreads();
  if (threadIdx.x == 0) {
    float mx = sc[0];
    for (int l = 1; l < LL; ++l) mx = fmaxf(mx, sc[l]);
    float sum = 0.f;
    for (int l = 0; l < LL; ++l) { float ev = expf(sc[l] - mx); sc[l] = ev; sum += ev; }
    float inv = 1.f / sum;
    for (int l = 0; l < LL; ++l) sc[l] *= inv;
  }
  __syncthreads();
  for (int k = threadIdx.x; k < HH; k += 256) {
    float a = 0.f;
    for (int l = 0; l < LL; ++l) a += sc[l] * enc[((size_t)b * LL + l) * HH + k];
    ctx_bf[b * HH + k] = (__bf16)a;
  }
}

// yc = tanh([h|ctx] @ concat_w^T + concat_b) -> yc_all[t]
__global__ __launch_bounds__(32)
void k_concat(const __bf16* __restrict__ h_bf, const __bf16* __restrict__ ctx_bf,
              const __bf16* __restrict__ cw_bf, const float* __restrict__ cb,
              __bf16* __restrict__ yc_t) {
  const int j = blockIdx.x;
  const int lane = threadIdx.x;
  f32x8 acc0 = (f32x8){}, acc1 = (f32x8){};
  // h half (K = 0..H-1)
#pragma unroll 4
  for (int kt = 0; kt < HH; kt += 32) {
    bf16x16 a0 = load_A(h_bf, HH, 0, kt);
    bf16x16 a1 = load_A(h_bf, HH, 16, kt);
    bf16x16 b  = load_Bt(cw_bf, 2 * HH, j * 16, kt);
    acc0 = wmma_bf16(a0, b, acc0);
    acc1 = wmma_bf16(a1, b, acc1);
  }
  // ctx half (K = H..2H-1)
#pragma unroll 4
  for (int kt = 0; kt < HH; kt += 32) {
    bf16x16 a0 = load_A(ctx_bf, HH, 0, kt);
    bf16x16 a1 = load_A(ctx_bf, HH, 16, kt);
    bf16x16 b  = load_Bt(cw_bf, 2 * HH, j * 16, HH + kt);
    acc0 = wmma_bf16(a0, b, acc0);
    acc1 = wmma_bf16(a1, b, acc1);
  }
  const int col = j * 16 + (lane & 15);
  const float bias = cb[col];
#pragma unroll
  for (int r = 0; r < 8; ++r) {
    int r0 = ((lane >> 4) << 3) + r;
    yc_t[r0 * HH + col]        = (__bf16)tanhf(acc0[r] + bias);
    yc_t[(16 + r0) * HH + col] = (__bf16)tanhf(acc1[r] + bias);
  }
}

// ---------------------------------------------------------------------------
// Batched output projection: logits[g] = yc_all[g] @ Wout^T + bout,
// g = t*32 + b over all 39 steps. Wout rows staged in LDS (read from L2 once);
// 8 waves per WG, one 16-col tile per wave, loop over all 78 M-tiles.
// ---------------------------------------------------------------------------
__global__ __launch_bounds__(256)
void k_logits_all(const __bf16* __restrict__ yc_all,   // [MROWS][HH]
                  const __bf16* __restrict__ wout_bf,  // [VV][HH]
                  const float* __restrict__ bout, float* __restrict__ dec) {
  __shared__ __bf16 sB[NROWS_LDS * HH];                // 128 KB
  const int n_base = blockIdx.x * NROWS_LDS;
  {
    const int nrows = min(NROWS_LDS, VV - n_base);
    const int chunks = nrows * (HH / 8);               // 16B chunks
    const bf16x8* src = (const bf16x8*)(wout_bf + (size_t)n_base * HH);
    bf16x8* dst = (bf16x8*)sB;
    for (int i = threadIdx.x; i < chunks; i += 256) dst[i] = src[i];
  }
  __syncthreads();

  const int w = threadIdx.x >> 5;
  const int lane = threadIdx.x & 31;
  const int tile = blockIdx.x * (NROWS_LDS / 16) + w;  // global 16-col tile
  if (tile * 16 >= VV) return;                          // uniform per wave

  const int col = tile * 16 + (lane & 15);
  const float bo = bout[col];
  const __bf16* bp = &sB[(w * 16 + (lane & 15)) * HH + ((lane >> 4) << 4)];

  for (int mt = 0; mt < MTILES; mt += 2) {
    f32x8 acc0 = (f32x8){}, acc1 = (f32x8){};
#pragma unroll 4
    for (int kt = 0; kt < HH; kt += 32) {
      bf16x16 a0 = load_A(yc_all, HH, mt * 16, kt);
      bf16x16 a1 = load_A(yc_all, HH, (mt + 1) * 16, kt);
      bf16x16 b  = *(const bf16x16*)(bp + kt);
      acc0 = wmma_bf16(a0, b, acc0);
      acc1 = wmma_bf16(a1, b, acc1);
    }
#pragma unroll
    for (int r = 0; r < 8; ++r) {
      int r0 = ((lane >> 4) << 3) + r;
      int g0 = mt * 16 + r0;
      int g1 = (mt + 1) * 16 + r0;
      dec[((size_t)(g0 & 31) * TT + ((g0 >> 5) + 1)) * VV + col] = acc0[r] + bo;
      dec[((size_t)(g1 & 31) * TT + ((g1 >> 5) + 1)) * VV + col] = acc1[r] + bo;
    }
  }
}

// symbol_outs[b][t+1] = argmax_v logits  (one block per (t,b), float index)
__global__ __launch_bounds__(256)
void k_argmax_all(float* __restrict__ out) {
  const int t = blockIdx.x >> 5;
  const int b = blockIdx.x & 31;
  const float* p = out + ((size_t)b * TT + (t + 1)) * VV;
  float best = -3.4e38f;
  int bi = 0;
  for (int k = threadIdx.x; k < VV; k += 256) {
    float v = p[k];
    if (v > best) { best = v; bi = k; }
  }
  __shared__ float sv[256];
  __shared__ int si[256];
  sv[threadIdx.x] = best; si[threadIdx.x] = bi;
  __syncthreads();
  for (int s = 128; s > 0; s >>= 1) {
    if (threadIdx.x < s) {
      float ov = sv[threadIdx.x + s]; int oi = si[threadIdx.x + s];
      if (ov > sv[threadIdx.x] || (ov == sv[threadIdx.x] && oi < si[threadIdx.x])) {
        sv[threadIdx.x] = ov; si[threadIdx.x] = oi;
      }
    }
    __syncthreads();
  }
  if (threadIdx.x == 0)
    out[(size_t)BB * TT * VV + (size_t)b * TT + (t + 1)] = (float)si[0];
}

// ---------------------------------------------------------------------------
extern "C" void kernel_launch(void* const* d_in, const int* in_sizes, int n_in,
                              void* d_out, int out_size, void* d_ws, size_t ws_size,
                              hipStream_t stream) {
  (void)in_sizes; (void)n_in; (void)out_size; (void)ws_size;
  const float* enc      = (const float*)d_in[0];
  const float* h0       = (const float*)d_in[1];
  const float* c0       = (const float*)d_in[2];
  const float* embed_w  = (const float*)d_in[3];
  const float* W_ih     = (const float*)d_in[4];
  const float* W_hh     = (const float*)d_in[5];
  const float* b_ih     = (const float*)d_in[6];
  const float* b_hh     = (const float*)d_in[7];
  const float* attn_w   = (const float*)d_in[8];
  const float* attn_b   = (const float*)d_in[9];
  const float* concat_w = (const float*)d_in[10];
  const float* concat_b = (const float*)d_in[11];
  const float* Wout     = (const float*)d_in[12];
  const float* bout     = (const float*)d_in[13];
  const int*   tok      = (const int*)d_in[14];
  float* out = (float*)d_out;

  char* ws = (char*)d_ws;
  size_t off = 0;
  auto carve = [&](size_t bytes) -> void* {
    void* p = ws + off;
    off = (off + bytes + 255) & ~((size_t)255);
    return p;
  };
  __bf16* wout_bf = (__bf16*)carve((size_t)VV * HH * 2);            // 30.7 MB
  __bf16* wcat_bf = (__bf16*)carve((size_t)4 * HH * 2 * HH * 2);    // 4 MB
  __bf16* attn_bf = (__bf16*)carve((size_t)HH * HH * 2);
  __bf16* cw_bf   = (__bf16*)carve((size_t)HH * 2 * HH * 2);
  __bf16* h_ping  = (__bf16*)carve((size_t)BB * HH * 2);
  __bf16* h_pong  = (__bf16*)carve((size_t)BB * HH * 2);
  __bf16* ctx_bf  = (__bf16*)carve((size_t)BB * HH * 2);
  __bf16* yc_all  = (__bf16*)carve((size_t)MROWS * HH * 2);         // 1.28 MB
  float*  c_st    = (float*)carve((size_t)BB * HH * 4);
  float*  e_f32   = (float*)carve((size_t)BB * HH * 4);

  // --- weight prep (bf16) ---
  k_build_wcat<<<(4 * HH * 2 * HH + 255) / 256, 256, 0, stream>>>(W_ih, W_hh, wcat_bf);
  k_cvt<<<1024, 256, 0, stream>>>(attn_w, attn_bf, HH * HH);
  k_cvt<<<1024, 256, 0, stream>>>(concat_w, cw_bf, HH * 2 * HH);
  k_cvt<<<4096, 256, 0, stream>>>(Wout, wout_bf, VV * HH);

  // --- state init + slot 0 (SOS) ---
  k_init<<<(BB * HH + 255) / 256, 256, 0, stream>>>(h0, c0, h_ping, c_st);
  k_sos<<<(VV + 7) / 8, 256, 0, stream>>>(embed_w, Wout, bout, out);

  // --- recurrence: 39 steps x 4 kernels; yc recorded for batched projection ---
  __bf16* hb[2] = {h_ping, h_pong};
  for (int t = 0; t < NSTEP; ++t) {
    const __bf16* h_in = hb[t & 1];
    __bf16* h_out = hb[(t + 1) & 1];
    k_gates_lstm<<<HH / 16, 32, 0, stream>>>(embed_w, tok, t, h_in, wcat_bf,
                                             b_ih, b_hh, c_st, h_out);
    k_attn_e<<<HH / 16, 32, 0, stream>>>(h_out, attn_bf, attn_b, e_f32);
    k_attn_ctx<<<BB, 256, 0, stream>>>(e_f32, enc, ctx_bf);
    k_concat<<<HH / 16, 32, 0, stream>>>(h_out, ctx_bf, cw_bf, concat_b,
                                         yc_all + (size_t)t * BB * HH);
  }

  // --- batched output projection + argmax over all steps ---
  k_logits_all<<<(VV + NROWS_LDS - 1) / NROWS_LDS, 256, 0, stream>>>(yc_all, wout_bf,
                                                                     bout, out);
  k_argmax_all<<<NSTEP * BB, 256, 0, stream>>>(out);
}
